// ExtendedANFIS_14035953123531
// MI455X (gfx1250) — matrix operand order repaired and checked by reading, not compiled, not run on hardware
//
#include <hip/hip_runtime.h>

// ExtendedANFIS forward for MI455X (gfx1250, wave32).
// out(16384x64) = normalize(outer-prod-of-Gaussian-memberships)(16384x2187) @ rules(2187x64)
//
// GEMM via v_wmma_f32_16x16x32_bf16 with a PERMUTED K order (we pack B ourselves, so any
// K permutation is legal): K' = (chunk c = l, pos = h) with rule r = h*81 + l, h in [0,27),
// pos 27..31 zero-padded (K_pad = 81*32 = 2592). Then each A tile is a rank-1 outer product
//   A[m][h] = phi[m][h] * plo[m][c]
// phi (27 vals, normalizer folded in) lives in 8 bf16-pair registers per lane (loop-
// invariant, pre-laid-out in WMMA-A lane order); plo is one broadcast dword per chunk.
// => per chunk per wave: 1 ds_load_b32 + 8 v_pk_mul_bf16 + 2 global_load_b128 + 1 wmma.
// No LDS staging, no in-loop barriers.

#define N_IN    7
#define N_MF    3
#define N_OUT   64
#define N_RULES 2187
#define NCHUNK  81            // one chunk per l = r % 81
#define EPSV    1e-6f

typedef __attribute__((ext_vector_type(16))) __bf16 v16bf;
typedef __attribute__((ext_vector_type(8)))  float  v8f;
typedef __attribute__((ext_vector_type(8)))  int    v8i;

static __device__ __forceinline__ unsigned int rnd_bf16(float f) {
  unsigned int u = __float_as_uint(f);
  return ((u + 0x7FFFu + ((u >> 16) & 1u)) >> 16) & 0xFFFFu;  // RNE bf16, low 16 bits
}
static __device__ __forceinline__ unsigned int f2bf_pack(float lo, float hi) {
  return rnd_bf16(lo) | (rnd_bf16(hi) << 16);
}
static __device__ __forceinline__ unsigned int f2bf_dup(float f) {
  const unsigned int b = rnd_bf16(f);
  return b | (b << 16);
}

// ---------------------------------------------------------------------------
// Kernel 1: pack rules into bf16 WMMA-B fragments in the PERMUTED K order.
// B tile for chunk c is 32x16 (K-pos x N): pos k < 27 -> rules[(k*81 + c)*64 + n], else 0.
// B layout used here (16-bit, same K order as A): lanes 0-15 hold K=0..15 (dword j -> K=2j,2j+1),
// lanes 16-31 hold K=16..31. 32 B per lane, contiguous:
// packed[(((c*4)+nt)*32 + lane)*8 + j]
// ---------------------------------------------------------------------------
__global__ void anfis_pack_rules(const float* __restrict__ rules,
                                 unsigned int* __restrict__ packed) {
  const int g    = blockIdx.x * blockDim.x + threadIdx.x;  // 0 .. 81*4*32-1
  const int lane = g & 31;
  const int nt   = (g >> 5) & 3;
  const int c    = g >> 7;                                 // chunk = l in [0,81)
  const int n    = (lane & 15) + nt * 16;
  const int khi  = (lane >= 16) ? 16 : 0;

  unsigned int outv[8];
#pragma unroll
  for (int j = 0; j < 8; ++j) {
    const int k0 = khi + 2 * j;          // K position (= h index)
    const int k1 = k0 + 1;
    const float v0 = (k0 < 27) ? rules[(k0 * 81 + c) * N_OUT + n] : 0.0f;
    const float v1 = (k1 < 27) ? rules[(k1 * 81 + c) * N_OUT + n] : 0.0f;
    outv[j] = f2bf_pack(v0, v1);
  }
#pragma unroll
  for (int j = 0; j < 8; ++j) packed[g * 8 + j] = outv[j];
}

// ---------------------------------------------------------------------------
// Kernel 2: per block: 16 batch rows, 128 threads = 4 waves; wave nt owns
// output columns [16*nt, 16*nt+16). 81 K-chunks, rank-1 A generation.
// ---------------------------------------------------------------------------
__global__ void __launch_bounds__(128)
anfis_main(const float* __restrict__ x, const float* __restrict__ mfs,
           const unsigned int* __restrict__ rules_packed,
           float* __restrict__ out) {
  __shared__ float        sh_memb[16][N_IN][N_MF + 1];  // +1 pad
  __shared__ float        sh_inv[16];
  __shared__ unsigned int sh_phi_pk[16][16];  // pair p = bf16{phi[2p], phi[2p+1]}, phi[27..31]=0
  __shared__ unsigned int sh_plo_dup[16][84]; // entry l = bf16 plo[m][l] duplicated in both halves

  const int tid  = threadIdx.x;
  const int row0 = blockIdx.x * 16;

  // --- 1) Gaussian memberships: 16 rows x 7 inputs, 3 MFs each -------------
  if (tid < 16 * N_IN) {
    const int row = tid / N_IN, inp = tid % N_IN;
    const float xv = x[(row0 + row) * N_IN + inp];
#pragma unroll
    for (int m = 0; m < N_MF; ++m) {
      const float mean = mfs[(inp * N_MF + m) * 2 + 0];
      const float sig  = fabsf(mfs[(inp * N_MF + m) * 2 + 1]) + EPSV;
      const float z    = (xv - mean) / sig;
      sh_memb[row][inp][m] = __expf(-0.5f * z * z);
    }
  }
  __syncthreads();

  // --- 2) closed-form normalizer: sum_r strengths = prod_i sum_m memb ------
  if (tid < 16) {
    float d = 1.0f;
#pragma unroll
    for (int inp = 0; inp < N_IN; ++inp)
      d *= sh_memb[tid][inp][0] + sh_memb[tid][inp][1] + sh_memb[tid][inp][2];
    sh_inv[tid] = 1.0f / (d + EPSV);
  }
  __syncthreads();

  // --- 3) factor tables in bf16-packed form --------------------------------
  // phi pairs: 16 rows x 16 pairs (h = 0..31, zeros for h >= 27), inv folded in
  for (int e = tid; e < 16 * 16; e += 128) {
    const int m = e >> 4, p = e & 15;
    float v0 = 0.0f, v1 = 0.0f;
    const int q0 = 2 * p, q1 = 2 * p + 1;
    if (q0 < 27)
      v0 = sh_memb[m][0][q0 / 9] * sh_memb[m][1][(q0 / 3) % 3] *
           sh_memb[m][2][q0 % 3] * sh_inv[m];
    if (q1 < 27)
      v1 = sh_memb[m][0][q1 / 9] * sh_memb[m][1][(q1 / 3) % 3] *
           sh_memb[m][2][q1 % 3] * sh_inv[m];
    sh_phi_pk[m][p] = f2bf_pack(v0, v1);
  }
  // plo: 16 rows x 81, duplicated bf16
  for (int e = tid; e < 16 * 81; e += 128) {
    const int m = e / 81, l = e % 81;
    const float v = sh_memb[m][3][l / 27] * sh_memb[m][4][(l / 9) % 3] *
                    sh_memb[m][5][(l / 3) % 3] * sh_memb[m][6][l % 3];
    sh_plo_dup[m][l] = f2bf_dup(v);
  }
  __syncthreads();

  const int lane = tid & 31;
  const int wave = tid >> 5;              // N-tile 0..3
  const int m    = lane & 15;
  const int hi4  = (lane >= 16) ? 4 : 0;

  // Loop-invariant phi fragment in WMMA-A lane order:
  // dword j holds bf16 pair for K = 2j + (lane>=16 ? 16:0)  => pair index
  unsigned int phi_pk[8];
#pragma unroll
  for (int j = 0; j < 8; ++j)
    phi_pk[j] = sh_phi_pk[m][j + 2 * hi4];

  const unsigned int* __restrict__ ploRow = sh_plo_dup[m];
  const v8i*          __restrict__ bp     = (const v8i*)rules_packed + (wave * 32 + lane);

  v8f acc = {};

#pragma unroll 3
  for (int c = 0; c < NCHUNK; ++c) {
    const unsigned int plo = ploRow[c];          // broadcast scalar of the rank-1 A tile

    unsigned int pr[8];
#pragma unroll
    for (int j = 0; j < 8; ++j)                  // A = phi (regs) * plo, packed bf16
      asm("v_pk_mul_bf16 %0, %1, %2" : "=v"(pr[j]) : "v"(phi_pk[j]), "v"(plo));

    v8i av;
#pragma unroll
    for (int j = 0; j < 8; ++j) av[j] = (int)pr[j];

    const v16bf a = __builtin_bit_cast(v16bf, av);
    const v16bf b = __builtin_bit_cast(v16bf, bp[c * 128]);
    acc = __builtin_amdgcn_wmma_f32_16x16x32_bf16(
        /*neg_a=*/false, a, /*neg_b=*/false, b,
        /*c_mod=*/(short)0, acc, /*reuse_a=*/false, /*reuse_b=*/false);
  }

  // D layout (f32 16x16): lane holds N = lane&15; VGPR v -> M = v + (lane>=16 ? 8:0)
  const int n     = (lane & 15) + wave * 16;
  const int mbase = (lane >= 16) ? 8 : 0;
#pragma unroll
  for (int v = 0; v < 8; ++v)
    out[(row0 + mbase + v) * N_OUT + n] = acc[v];
}

// ---------------------------------------------------------------------------
extern "C" void kernel_launch(void* const* d_in, const int* in_sizes, int n_in,
                              void* d_out, int out_size, void* d_ws, size_t ws_size,
                              hipStream_t stream) {
  (void)in_sizes; (void)n_in; (void)out_size; (void)ws_size;
  const float* x     = (const float*)d_in[0];   // (16384, 7)
  const float* mfs   = (const float*)d_in[1];   // (7, 3, 2)
  const float* rules = (const float*)d_in[2];   // (2187, 64)
  float*       out   = (float*)d_out;           // (16384, 64)
  unsigned int* packed = (unsigned int*)d_ws;   // needs 81*4*32*32 = 331,776 B

  // 1) pack rules into bf16 WMMA-B fragments, permuted K order (81 chunks x 4 ntiles)
  anfis_pack_rules<<<NCHUNK, 128, 0, stream>>>(rules, packed);

  // 2) fused memberships + rank-1 strength generation + WMMA GEMM
  anfis_main<<<16384 / 16, 128, 0, stream>>>(x, mfs, packed, out);
}
